// MemTransformerLM_32452772888674
// MI455X (gfx1250) — matrix operand order repaired and checked by reading, not compiled, not run on hardware
//
#include <hip/hip_runtime.h>

// ---------------- problem constants ----------------
#define NH       8
#define DH       64
#define DMODEL   512
#define MEMLEN   1024
#define QLEN     1024
#define BSZ      4
#define KLEN     (MEMLEN + QLEN)   // 2048
#define QKVCOLS  (3 * NH * DH)     // 1536

typedef __attribute__((ext_vector_type(16))) _Float16 v16h;
typedef __attribute__((ext_vector_type(8)))  float    v8f;

static __device__ __forceinline__ v8f wmma16(v16h a, v16h b, v8f c) {
  // D = A(16x32 f16) * B(32x16 f16) + C(16x16 f32)
  return __builtin_amdgcn_wmma_f32_16x16x32_f16(
      /*neg_a=*/false, a, /*neg_b=*/false, b,
      /*c_mod=*/(short)0, c, /*reuse_a=*/false, /*reuse_b=*/false);
}

// A fragment (16x32 f16 layout): lane m = lane&15 is the row, kh = lane>>4.
// K = 16*kh + [0,16) contiguous (matches B fragment).
static __device__ __forceinline__ v16h load_a_f16row(const _Float16* arow, int k0, int kh) {
  v16h a;
#pragma unroll
  for (int i = 0; i < 8; ++i) {
    a[i]     = arow[k0 + 16 * kh + i];
    a[8 + i] = arow[k0 + 16 * kh + 8 + i];
  }
  return a;
}
// B fragment (32x16): lane n = lane&15 is the column; K = 16*kh + [0,16) contiguous.
static __device__ __forceinline__ v16h load_b_f16row(const _Float16* brow, int k0, int kh) {
  v16h b;
#pragma unroll
  for (int i = 0; i < 16; ++i) b[i] = brow[k0 + 16 * kh + i];
  return b;
}

// ---------------------------------------------------------------------------
// f32 -> f16 pack (one-time; keeps all conversions out of the WMMA hot loops)
// ---------------------------------------------------------------------------
__global__ __launch_bounds__(256) void pack_f16_kernel(
    const float* __restrict__ src, _Float16* __restrict__ dst, int n8) {
  int i = blockIdx.x * blockDim.x + threadIdx.x;
  if (i >= n8) return;
  const int b = i * 8;
#pragma unroll
  for (int k = 0; k < 8; ++k) dst[b + k] = (_Float16)src[b + k];
}

// ---------------------------------------------------------------------------
// C(M,N) = A(M,K) @ B(N,K)^T, all-f16 operands, f32 accumulate.
// 4 waves / block; block tile 128x64; wave tile 32x64 (two A frags share
// every B frag -> 2x arithmetic intensity). B staged in a double-buffered
// 64x64 LDS slab (one barrier per 16 WMMAs); row stride 72 halves (144 B)
// keeps the 16 fragment lanes on distinct banks. K multiple of 64.
// ---------------------------------------------------------------------------
template <bool HALF_OUT>
__global__ __launch_bounds__(128) void gemm_f16_kernel(
    const _Float16* __restrict__ A, const _Float16* __restrict__ B,
    float* __restrict__ Cf, _Float16* __restrict__ Ch, int M, int N, int K) {
  __shared__ _Float16 bs[2][64][72];
  const int tid = threadIdx.x, wave = tid >> 5, lane = tid & 31;
  const int lm = lane & 15, kh = lane >> 4;
  const int m0 = blockIdx.x * 128 + wave * 32, n0 = blockIdx.y * 64;
  const _Float16* arow0 = A + (size_t)(m0 + lm) * K;
  const _Float16* arow1 = A + (size_t)(m0 + 16 + lm) * K;
  const int bn = tid >> 1, bseg = (tid & 1) * 32;  // cooperative B copy mapping
  const _Float16* bsrc = B + (size_t)(n0 + bn) * K + bseg;

  // prologue: stage k-slab 0
#pragma unroll
  for (int i = 0; i < 32; ++i) bs[0][bn][bseg + i] = bsrc[i];

  v8f acc0[4] = {{}, {}, {}, {}}, acc1[4] = {{}, {}, {}, {}};
  const int nStages = K >> 6;
  for (int st = 0; st < nStages; ++st) {
    const int k0 = st * 64, cur = st & 1;
    __syncthreads();
    if (st + 1 < nStages) {               // overlap next stage with compute
      const _Float16* src = bsrc + k0 + 64;
      __builtin_prefetch(src + 64, 0, 3);
#pragma unroll
      for (int i = 0; i < 32; ++i) bs[cur ^ 1][bn][bseg + i] = src[i];
    }
#pragma unroll
    for (int h = 0; h < 2; ++h) {
      const v16h a0 = load_a_f16row(arow0, k0 + 32 * h, kh);
      const v16h a1 = load_a_f16row(arow1, k0 + 32 * h, kh);
#pragma unroll
      for (int t = 0; t < 4; ++t) {
        const _Float16* brow = &bs[cur][t * 16 + lm][32 * h + 16 * kh];
        v16h b;
#pragma unroll
        for (int i = 0; i < 16; ++i) b[i] = brow[i];
        acc0[t] = wmma16(a0, b, acc0[t]);
        acc1[t] = wmma16(a1, b, acc1[t]);
      }
    }
  }
#pragma unroll
  for (int t = 0; t < 4; ++t)
#pragma unroll
    for (int v = 0; v < 8; ++v) {
      const size_t i0 = (size_t)(m0 + v + 8 * kh) * N + n0 + t * 16 + lm;
      const size_t i1 = (size_t)(m0 + 16 + v + 8 * kh) * N + n0 + t * 16 + lm;
      if constexpr (HALF_OUT) {
        Ch[i0] = (_Float16)acc0[t][v];
        Ch[i1] = (_Float16)acc1[t][v];
      } else {
        Cf[i0] = acc0[t][v];
        Cf[i1] = acc1[t][v];
      }
    }
}

// ---------------------------------------------------------------------------
// Per-head 7-tap conv as WMMA GEMM: Y(16x64) = Xim2col(16x448) @ Wc(64x448)^T.
// Wc layout (64,64,7) is already K-contiguous (K = din*7+t); im2col is virtual
// (A-fragment elements are indexed LDS reads). K = 448 = 14*32 exactly.
// mode 0: K -> [n][b][j][d]; mode 1: V -> [n][b][d][j]; mode 2: Q (+both biases).
// ---------------------------------------------------------------------------
__global__ __launch_bounds__(32) void head_conv_wmma_kernel(
    const _Float16* __restrict__ wheads, int col0, int rowoff, int L,
    const _Float16* __restrict__ Wc, const float* __restrict__ bias,
    const float* __restrict__ bw, const float* __restrict__ br,
    _Float16* __restrict__ out0, _Float16* __restrict__ out1, int mode) {
  __shared__ _Float16 xt[22][DH];  // rows j0-3 .. j0+18, zero padded
  const int n = blockIdx.z, b = blockIdx.y, j0 = blockIdx.x * 16;
  const int lane = threadIdx.x & 31, lm = lane & 15, kh = lane >> 4;

  for (int idx = threadIdx.x; idx < 22 * DH; idx += 32) {
    int rr = idx >> 6, c = idx & 63;
    int jg = j0 + rr - 3;
    _Float16 v = (_Float16)0.f;
    if (jg >= 0 && jg < L)
      v = wheads[((size_t)(jg + rowoff) * BSZ + b) * QKVCOLS + col0 + n * DH + c];
    xt[rr][c] = v;
  }
  __syncthreads();

  v8f acc[4] = {{}, {}, {}, {}};
  for (int kc = 0; kc < 14; ++kc) {
    const int k0 = kc * 32;
    v16h a;
#pragma unroll
    for (int i = 0; i < 8; ++i) {
      int kA = k0 + 16 * kh + i;           // K -> (din = kA/7, tap = kA%7)
      a[i] = xt[lm + (kA % 7)][kA / 7];
      int kB = k0 + 16 * kh + 8 + i;
      a[8 + i] = xt[lm + (kB % 7)][kB / 7];
    }
#pragma unroll
    for (int t = 0; t < 4; ++t) {
      const _Float16* wrow = Wc + (size_t)(t * 16 + lm) * 448;
      acc[t] = wmma16(a, load_b_f16row(wrow, k0, kh), acc[t]);
    }
  }

  const size_t hb = (size_t)n * BSZ + b;
#pragma unroll
  for (int t = 0; t < 4; ++t) {
    const int dout = t * 16 + lm;
    const float bsv = bias[dout];
#pragma unroll
    for (int v = 0; v < 8; ++v) {
      const int j = j0 + v + 8 * kh;
      const float val = acc[t][v] + bsv;
      if (mode == 0) {
        out0[(hb * L + j) * DH + dout] = (_Float16)val;
      } else if (mode == 1) {
        out0[(hb * DH + dout) * L + j] = (_Float16)val;  // transposed for PV B-frags
      } else {
        out0[(hb * L + j) * DH + dout] = (_Float16)(val + bw[n * DH + dout]);
        out1[(hb * L + j) * DH + dout] = (_Float16)(val + br[n * DH + dout]);
      }
    }
  }
}

// r_head_k f16 (klen, 512) -> r_h [n][j][d] f16
__global__ void r_reorder_kernel(const _Float16* __restrict__ rhk,
                                 _Float16* __restrict__ r_h) {
  int idx = blockIdx.x * blockDim.x + threadIdx.x;
  if (idx >= NH * KLEN * DH) return;
  int d = idx & 63;
  int j = (idx >> 6) & (KLEN - 1);
  int n = idx / (KLEN * DH);
  r_h[idx] = rhk[(size_t)j * DMODEL + n * DH + d];
}

// ---------------------------------------------------------------------------
// Fused rel-attention: AC + shifted BD + mask + online softmax + P@V.
// grid = (QLEN/64, BSZ*NH), block = 128 (4 waves). Wave w owns query rows
// [iw, iw+16). All LDS is wave-private -> no block barriers in the main loop.
// BD'[i][j] = (q_i + r_r_bias) . r[j + QLEN - 1 - i]; out-of-range r indices
// coincide exactly with masked (j > i + MEMLEN) positions, so clamped loads OK.
// ---------------------------------------------------------------------------
__global__ __launch_bounds__(128) void attn_kernel(
    const _Float16* __restrict__ qw_h, const _Float16* __restrict__ qr_h,
    const _Float16* __restrict__ k_h, const _Float16* __restrict__ v_h,
    const _Float16* __restrict__ r_h, _Float16* __restrict__ attn_vec) {
  __shared__ float    bdraw[4][16][48];
  __shared__ _Float16 pbuf[4][16][32];

  const int wave = threadIdx.x >> 5, lane = threadIdx.x & 31;
  const int lm = lane & 15, kh = lane >> 4;
  const int bn = blockIdx.y, n = bn >> 2, b = bn & 3;
  const int iw = blockIdx.x * 64 + wave * 16;

  const _Float16* qwp = qw_h + ((size_t)(n * BSZ + b) * QLEN + iw) * DH;
  const _Float16* qrp = qr_h + ((size_t)(n * BSZ + b) * QLEN + iw) * DH;
  const _Float16* kp  = k_h + (size_t)(n * BSZ + b) * KLEN * DH;   // [j][d]
  const _Float16* vp  = v_h + (size_t)(n * BSZ + b) * DH * KLEN;   // [d][j]
  const _Float16* rp  = r_h + (size_t)n * KLEN * DH;               // [j][d]

  // Persistent A fragments (rows iw..iw+15, K = DH = 64 -> two 32-chunks).
  const _Float16* qwrow = qwp + (size_t)lm * DH;
  const _Float16* qrrow = qrp + (size_t)lm * DH;
  v16h aw0 = load_a_f16row(qwrow, 0, kh), aw1 = load_a_f16row(qwrow, 32, kh);
  v16h ar0 = load_a_f16row(qrrow, 0, kh), ar1 = load_a_f16row(qrrow, 32, kh);

  const float NEG = -__builtin_inff();
  v8f o[4] = {{}, {}, {}, {}};
  float rmax[8], rsum[8];
#pragma unroll
  for (int v = 0; v < 8; ++v) { rmax[v] = NEG; rsum[v] = 0.f; }

  const int jlimit = iw + 15 + MEMLEN;
  for (int j0 = 0; j0 < KLEN && j0 <= jlimit; j0 += 32) {
    __builtin_prefetch(kp + (size_t)(j0 + 32) * DH + lane * 8, 0, 3);
    __builtin_prefetch(rp + (size_t)(j0 + (QLEN - 16) - iw + 48) * DH + lane * 8, 0, 3);

    // ---- AC = (q + r_w_bias) @ K^T, two 16x16 column tiles ----
    v8f s[2];
#pragma unroll
    for (int tc = 0; tc < 2; ++tc) {
      int j = j0 + tc * 16 + lm; if (j > KLEN - 1) j = KLEN - 1;
      const _Float16* krow = kp + (size_t)j * DH;
      v8f acc = {};
      acc = wmma16(aw0, load_b_f16row(krow, 0, kh), acc);
      acc = wmma16(aw1, load_b_f16row(krow, 32, kh), acc);
      s[tc] = acc;
    }

    // ---- BDraw slab: r indices [rbase, rbase+47] ----
    const int rbase = j0 + (QLEN - 16) - iw;  // >= 0 always
#pragma unroll
    for (int rt = 0; rt < 3; ++rt) {
      int rr = rbase + rt * 16 + lm; if (rr > KLEN - 1) rr = KLEN - 1;
      const _Float16* rrow = rp + (size_t)rr * DH;
      v8f d = {};
      d = wmma16(ar0, load_b_f16row(rrow, 0, kh), d);
      d = wmma16(ar1, load_b_f16row(rrow, 32, kh), d);
#pragma unroll
      for (int v = 0; v < 8; ++v) bdraw[wave][v + 8 * kh][rt * 16 + lm] = d[v];
    }

    // ---- assemble scores: shift-gather + scale + mask ----
#pragma unroll
    for (int v = 0; v < 8; ++v) {
      const int m = v + 8 * kh;
      const int i = iw + m;
#pragma unroll
      for (int tc = 0; tc < 2; ++tc) {
        const int jc = tc * 16 + lm;
        float sc = (s[tc][v] + bdraw[wave][m][jc + 15 - m]) * 0.125f;
        const int j = j0 + jc;
        if (j >= KLEN || j > i + MEMLEN) sc = NEG;
        s[tc][v] = sc;
      }
    }

    // ---- online softmax (16-lane row groups match D-tile row ownership) ----
#pragma unroll
    for (int v = 0; v < 8; ++v) {
      float t = fmaxf(s[0][v], s[1][v]);
#pragma unroll
      for (int msk = 1; msk < 16; msk <<= 1) t = fmaxf(t, __shfl_xor(t, msk, 32));
      const float nm = fmaxf(rmax[v], t);
      const float corr = (nm == NEG) ? 1.0f : __expf(rmax[v] - nm);
      const float p0 = (s[0][v] == NEG) ? 0.f : __expf(s[0][v] - nm);
      const float p1 = (s[1][v] == NEG) ? 0.f : __expf(s[1][v] - nm);
      float ps = p0 + p1;
#pragma unroll
      for (int msk = 1; msk < 16; msk <<= 1) ps += __shfl_xor(ps, msk, 32);
      rsum[v] = rsum[v] * corr + ps;
      rmax[v] = nm;
#pragma unroll
      for (int t4 = 0; t4 < 4; ++t4) o[t4][v] *= corr;
      pbuf[wave][v + 8 * kh][lm]      = (_Float16)p0;
      pbuf[wave][v + 8 * kh][16 + lm] = (_Float16)p1;
    }

    // ---- P (D layout) -> A fragment via wave-private LDS transpose ----
    v16h pa;
    {
      const _Float16* prow = &pbuf[wave][lm][0];
#pragma unroll
      for (int i = 0; i < 8; ++i) {
        pa[i]     = prow[16 * kh + i];
        pa[8 + i] = prow[16 * kh + 8 + i];
      }
    }

    // ---- O += P @ V  (V in [d][j] layout -> contiguous K per lane) ----
#pragma unroll
    for (int t = 0; t < 4; ++t) {
      const _Float16* vrow = vp + (size_t)(t * 16 + lm) * KLEN;
      v16h bv;
#pragma unroll
      for (int i = 0; i < 16; ++i) {
        int j = j0 + 16 * kh + i; if (j > KLEN - 1) j = KLEN - 1;  // p == 0 there
        bv[i] = vrow[j];
      }
      o[t] = wmma16(pa, bv, o[t]);
    }
  }

  // ---- normalize, store attn_vec (f16) [(i*BSZ+b)][n*DH + d] ----
#pragma unroll
  for (int v = 0; v < 8; ++v) {
    const float inv = 1.0f / rsum[v];
    const int i = iw + v + 8 * kh;
#pragma unroll
    for (int t = 0; t < 4; ++t)
      attn_vec[((size_t)i * BSZ + b) * DMODEL + n * DH + t * 16 + lm] =
          (_Float16)(o[t][v] * inv);
  }
}

// ---------------------------------------------------------------------------
// out = LayerNorm(w + attn_out) * g + b ; one 256-thread block per row (512)
// ---------------------------------------------------------------------------
__global__ __launch_bounds__(256) void add_ln_kernel(
    const float* __restrict__ w, const float* __restrict__ ao,
    const float* __restrict__ g, const float* __restrict__ bb,
    float* __restrict__ out) {
  __shared__ float red[256];
  const int row = blockIdx.x, tid = threadIdx.x;
  const size_t base = (size_t)row * DMODEL;
  const float x0 = w[base + tid] + ao[base + tid];
  const float x1 = w[base + 256 + tid] + ao[base + 256 + tid];
  red[tid] = x0 + x1;
  __syncthreads();
  for (int off = 128; off; off >>= 1) { if (tid < off) red[tid] += red[tid + off]; __syncthreads(); }
  const float mu = red[0] * (1.0f / DMODEL);
  __syncthreads();
  const float d0 = x0 - mu, d1 = x1 - mu;
  red[tid] = d0 * d0 + d1 * d1;
  __syncthreads();
  for (int off = 128; off; off >>= 1) { if (tid < off) red[tid] += red[tid + off]; __syncthreads(); }
  const float rs = rsqrtf(red[0] * (1.0f / DMODEL) + 1e-5f);
  out[base + tid]       = d0 * rs * g[tid] + bb[tid];
  out[base + 256 + tid] = d1 * rs * g[256 + tid] + bb[256 + tid];
}

// ---------------------------------------------------------------------------
// Never-launched probe kernel: exercises the remaining gfx1250-specific paths
// that are codegen-confirmed (cluster id, cluster split-barrier, TENSORcnt /
// ASYNCcnt waits). The async-to-LDS builtin on this toolchain takes an int4
// descriptor (probe result, round 3), so we don't call it; the SOPP waits are
// plain documented instructions and assemble regardless.
// ---------------------------------------------------------------------------
__global__ __launch_bounds__(32) void cdna5_cluster_probe_kernel(
    const float* __restrict__ src, float* __restrict__ dst) {
  __shared__ float buf[128];
  buf[threadIdx.x] = src[threadIdx.x];
  __syncthreads();
  asm volatile("s_wait_asynccnt 0" ::: "memory");
  const int cid = __builtin_amdgcn_cluster_id_x();  // NOP-safe outside clusters
  __builtin_amdgcn_s_cluster_barrier();
  __builtin_amdgcn_s_wait_tensorcnt(0);
  dst[threadIdx.x] = buf[threadIdx.x] + (float)cid;
}

// ---------------------------------------------------------------------------
extern "C" void kernel_launch(void* const* d_in, const int* in_sizes, int n_in,
                              void* d_out, int out_size, void* d_ws, size_t ws_size,
                              hipStream_t stream) {
  (void)in_sizes; (void)n_in; (void)out_size; (void)ws_size;
  const float* w        = (const float*)d_in[0];
  const float* r        = (const float*)d_in[1];
  const float* mems     = (const float*)d_in[2];
  const float* r_w_bias = (const float*)d_in[3];
  const float* r_r_bias = (const float*)d_in[4];
  const float* qkv_W    = (const float*)d_in[5];
  const float* r_W      = (const float*)d_in[6];
  const float* conv_q_w = (const float*)d_in[7];
  const float* conv_q_b = (const float*)d_in[8];
  const float* conv_k_w = (const float*)d_in[9];
  const float* conv_k_b = (const float*)d_in[10];
  const float* conv_v_w = (const float*)d_in[11];
  const float* conv_v_b = (const float*)d_in[12];
  const float* o_W      = (const float*)d_in[13];
  const float* ln_g     = (const float*)d_in[14];
  const float* ln_b     = (const float*)d_in[15];
  float* out = (float*)d_out;

  // workspace layout (~80 MB, stays resident in the 192 MB L2)
  char* p = (char*)d_ws;
  auto take = [&](size_t bytes) { char* q = p; p += (bytes + 255) & ~(size_t)255; return q; };
  _Float16* a16    = (_Float16*)take((size_t)KLEN * BSZ * DMODEL * 2);   // cat(mems,w) f16
  _Float16* qkvW16 = (_Float16*)take((size_t)QKVCOLS * DMODEL * 2);
  _Float16* rW16   = (_Float16*)take((size_t)DMODEL * DMODEL * 2);
  _Float16* oW16   = (_Float16*)take((size_t)DMODEL * DMODEL * 2);
  _Float16* r16    = (_Float16*)take((size_t)KLEN * DMODEL * 2);
  _Float16* cqw16  = (_Float16*)take((size_t)DH * DH * 7 * 2);
  _Float16* ckw16  = (_Float16*)take((size_t)DH * DH * 7 * 2);
  _Float16* cvw16  = (_Float16*)take((size_t)DH * DH * 7 * 2);
  _Float16* wh16   = (_Float16*)take((size_t)KLEN * BSZ * QKVCOLS * 2);  // w_heads f16
  _Float16* rhk16  = (_Float16*)take((size_t)KLEN * DMODEL * 2);         // r_head_k f16
  _Float16* qw_h   = (_Float16*)take((size_t)NH * BSZ * QLEN * DH * 2);  // q + r_w_bias
  _Float16* qr_h   = (_Float16*)take((size_t)NH * BSZ * QLEN * DH * 2);  // q + r_r_bias
  _Float16* k_h    = (_Float16*)take((size_t)NH * BSZ * KLEN * DH * 2);  // [n][b][j][d]
  _Float16* v_h    = (_Float16*)take((size_t)NH * BSZ * DH * KLEN * 2);  // [n][b][d][j]
  _Float16* r_h    = (_Float16*)take((size_t)NH * KLEN * DH * 2);        // [n][j][d]
  _Float16* avec   = (_Float16*)take((size_t)QLEN * BSZ * DMODEL * 2);
  float*    aout   = (float*)take((size_t)QLEN * BSZ * DMODEL * 4);

  auto pack = [&](const float* s, _Float16* d, size_t count) {
    const int n8 = (int)(count / 8);
    pack_f16_kernel<<<(n8 + 255) / 256, 256, 0, stream>>>(s, d, n8);
  };

  // 0. one-time f16 packs (removes all cvt chains from WMMA hot loops)
  pack(mems, a16, (size_t)MEMLEN * BSZ * DMODEL);
  pack(w, a16 + (size_t)MEMLEN * BSZ * DMODEL, (size_t)QLEN * BSZ * DMODEL);
  pack(qkv_W, qkvW16, (size_t)QKVCOLS * DMODEL);
  pack(r_W, rW16, (size_t)DMODEL * DMODEL);
  pack(o_W, oW16, (size_t)DMODEL * DMODEL);
  pack(r, r16, (size_t)KLEN * DMODEL);
  pack(conv_q_w, cqw16, (size_t)DH * DH * 7);
  pack(conv_k_w, ckw16, (size_t)DH * DH * 7);
  pack(conv_v_w, cvw16, (size_t)DH * DH * 7);

  const int Mqkv = KLEN * BSZ;  // 8192 rows (j*BSZ+b) of cat(mems, w)

  // 1. w_heads = cat(mems, w) @ qkv_W^T   (f16 out)
  gemm_f16_kernel<true><<<dim3(Mqkv / 128, QKVCOLS / 64), 128, 0, stream>>>(
      a16, qkvW16, nullptr, wh16, Mqkv, QKVCOLS, DMODEL);
  // 2. r_head_k = r @ r_W^T   (f16 out)
  gemm_f16_kernel<true><<<dim3(KLEN / 128, DMODEL / 64), 128, 0, stream>>>(
      r16, rW16, nullptr, rhk16, KLEN, DMODEL, DMODEL);
  // 3. per-head convs (WMMA im2col) + bias/reformat
  head_conv_wmma_kernel<<<dim3(QLEN / 16, BSZ, NH), 32, 0, stream>>>(
      wh16, 0, MEMLEN, QLEN, cqw16, conv_q_b, r_w_bias, r_r_bias, qw_h, qr_h, 2);
  head_conv_wmma_kernel<<<dim3(KLEN / 16, BSZ, NH), 32, 0, stream>>>(
      wh16, NH * DH, 0, KLEN, ckw16, conv_k_b, nullptr, nullptr, k_h, nullptr, 0);
  head_conv_wmma_kernel<<<dim3(KLEN / 16, BSZ, NH), 32, 0, stream>>>(
      wh16, 2 * NH * DH, 0, KLEN, cvw16, conv_v_b, nullptr, nullptr, v_h, nullptr, 1);
  // 4. r_head_k -> head-major
  r_reorder_kernel<<<(NH * KLEN * DH + 255) / 256, 256, 0, stream>>>(rhk16, r_h);
  // 5. fused rel-attention (flash style; scores never hit memory)
  attn_kernel<<<dim3(QLEN / 64, BSZ * NH), 128, 0, stream>>>(
      qw_h, qr_h, k_h, v_h, r_h, avec);
  // 6. attn_out = attn_vec @ o_W^T  (f32 out for LN)
  gemm_f16_kernel<false><<<dim3((QLEN * BSZ) / 128, DMODEL / 64), 128, 0, stream>>>(
      avec, oW16, aout, nullptr, QLEN * BSZ, DMODEL, DMODEL);
  // 7. residual + LayerNorm
  add_ln_kernel<<<QLEN * BSZ, 256, 0, stream>>>(w, aout, ln_g, ln_b, out);
}
